// MessageLayer_48241072668742
// MI455X (gfx1250) — matrix operand order repaired and checked by reading, not compiled, not run on hardware
//
#include <hip/hip_runtime.h>
#include <hip/hip_bf16.h>
#include <math.h>

typedef __attribute__((ext_vector_type(16))) __bf16 v16bf;
typedef __attribute__((ext_vector_type(8)))  __bf16 v8bf;
typedef __attribute__((ext_vector_type(8)))  float  v8f;

#define NN   50000
#define MM   1250000
#define DD   64
#define DINC 192
#define NT   78125            // MM / 16 edge tiles
#define WPB  8                // waves per block (256 threads, wave32)
#define NB1  ((NT + WPB - 1) / WPB)   // 9766 blocks for k_gemm

// ---------------------------------------------------------------------------
// Kernel 0: pack W = [Wf | Wc] transposed into bf16  Wt[n][k], n<64->Wf, else Wc
// ---------------------------------------------------------------------------
__global__ void k_prep(const float* __restrict__ Wf, const float* __restrict__ Wc,
                       __bf16* __restrict__ Wt) {
  for (int i = threadIdx.x; i < 128 * DINC; i += 256) {
    int n = i / DINC, k = i % DINC;
    float w = (n < 64) ? Wf[k * 64 + n] : Wc[k * 64 + (n - 64)];
    Wt[i] = (__bf16)w;
  }
}

// ---------------------------------------------------------------------------
// Kernel 1: edge GEMM via bf16 WMMA + deterministic BN-stat partial sums
//   one wave = one 16-edge tile; z = total_fea @ [Wf|Wc]  (bias cancels in BN)
// ---------------------------------------------------------------------------
__global__ void __launch_bounds__(256)
k_gemm(const float* __restrict__ atom, const float* __restrict__ bond,
       const int* __restrict__ selfi, const int* __restrict__ nbri,
       const __bf16* __restrict__ Wt,
       __bf16* __restrict__ zf, __bf16* __restrict__ zc,
       float* __restrict__ partial) {
  __shared__ __bf16 astage[WPB * 16 * DINC];   // 48 KB: per-wave A staging
  __shared__ float  wpart[WPB * 256];          // 8 KB: per-wave stat partials

  const int tid  = threadIdx.x;
  const int wv   = tid >> 5;
  const int lane = tid & 31;
  const int m    = lane & 15;     // row within tile (A) / column within N-tile (B,D)
  const int h    = lane >> 4;     // K-half selector

  for (int i = tid; i < WPB * 256; i += 256) wpart[i] = 0.0f;
  __syncthreads();

  const int tile = blockIdx.x * WPB + wv;
  if (tile < NT) {
    const int E0 = tile * 16;
    __bf16* st = astage + wv * 16 * DINC;

    // prefetch next tile's bond stream (global_prefetch_b8)
    if (E0 + 16 * WPB < MM)
      __builtin_prefetch(bond + (size_t)(E0 + 16 * WPB) * 64 + lane * 2, 0, 0);

    // stage 16 edges x 192 cols of gathered fp32 features as bf16 (768 float4s)
    for (int j = lane; j < 768; j += 32) {
      int r = j / 48, q = j % 48;
      int e = E0 + r;
      float4 v;
      if (q < 16)      v = ((const float4*)(atom + (size_t)selfi[e] * 64))[q];
      else if (q < 32) v = ((const float4*)(atom + (size_t)nbri[e] * 64))[q - 16];
      else             v = ((const float4*)(bond + (size_t)e * 64))[q - 32];
      __bf16* d = st + r * DINC + q * 4;
      d[0] = (__bf16)v.x; d[1] = (__bf16)v.y; d[2] = (__bf16)v.z; d[3] = (__bf16)v.w;
    }

    union AB { v16bf v; v8bf p[2]; };
    // A fragments: ISA 16-bit 16x32 A layout -> elems 0-7 @ col k*32+h*8,
    // elems 8-15 @ col k*32+16+h*8 (two 16B LDS loads each)
    AB a[6];
#pragma unroll
    for (int k = 0; k < 6; ++k) {
      const __bf16* p = st + m * DINC + k * 32 + h * 8;
      a[k].p[0] = *(const v8bf*)p;
      a[k].p[1] = *(const v8bf*)(p + 16);
    }

    v8f acc[8] = {};
#pragma unroll
    for (int t = 0; t < 8; ++t) {
      const __bf16* q0 = Wt + (size_t)(t * 16 + m) * DINC + h * 8;
#pragma unroll
      for (int k = 0; k < 6; ++k) {
        AB b;                                   // B: K h*8..+7 and 16+h*8..+7 per lane
        const __bf16* q = q0 + k * 32;
        b.p[0] = *(const v8bf*)q;
        b.p[1] = *(const v8bf*)(q + 16);
        acc[t] = __builtin_amdgcn_wmma_f32_16x16x32_bf16(
            false, a[k].v, false, b.v, (short)0, acc[t], false, false);
      }
    }

    // per-column partial sums (deterministic: shfl pair-reduce, lanes 0-15 write)
#pragma unroll
    for (int t = 0; t < 8; ++t) {
      float s1 = 0.f, s2 = 0.f;
#pragma unroll
      for (int v = 0; v < 8; ++v) { float x = acc[t][v]; s1 += x; s2 += x * x; }
      s1 += __shfl_xor(s1, 16, 32);
      s2 += __shfl_xor(s2, 16, 32);
      int c = t * 16 + m;
      if (h == 0) { wpart[wv * 256 + c] = s1; wpart[wv * 256 + 128 + c] = s2; }
      // store z as bf16 (D layout: row = v + 8*h, col = t*16 + m)
      __bf16* dst = (t < 4) ? (zf + c) : (zc + (c - 64));
#pragma unroll
      for (int v = 0; v < 8; ++v)
        dst[(size_t)(E0 + v + 8 * h) * 64] = (__bf16)acc[t][v];
    }
  }
  __syncthreads();
  float s = 0.f;
#pragma unroll
  for (int w = 0; w < WPB; ++w) s += wpart[w * 256 + tid];   // fixed order
  partial[(size_t)blockIdx.x * 256 + tid] = s;
}

// ---------------------------------------------------------------------------
// Kernels 2a/2b: reduce BN stats, emit per-column scale/shift
// ---------------------------------------------------------------------------
__global__ void k_red1(const float* __restrict__ partial, float* __restrict__ partial2) {
  int tid = threadIdx.x;
  float s = 0.f;
  for (int r = blockIdx.x; r < NB1; r += 256) s += partial[(size_t)r * 256 + tid];
  partial2[(size_t)blockIdx.x * 256 + tid] = s;
}

__global__ void k_red2(const float* __restrict__ partial2,
                       const float* __restrict__ gf, const float* __restrict__ betaf,
                       const float* __restrict__ gc, const float* __restrict__ betac,
                       float* __restrict__ bnp) {
  __shared__ float s[256];
  int tid = threadIdx.x;
  float v = 0.f;
  for (int r = 0; r < 256; ++r) v += partial2[r * 256 + tid];
  s[tid] = v;
  __syncthreads();
  if (tid < 128) {
    const float invM = 1.0f / (float)MM;
    float mu  = s[tid] * invM;
    float var = fmaxf(s[tid + 128] * invM - mu * mu, 0.0f);
    float g   = (tid < 64) ? gf[tid]    : gc[tid - 64];
    float be  = (tid < 64) ? betaf[tid] : betac[tid - 64];
    float sc  = g * rsqrtf(var + 1e-5f);
    bnp[tid]       = sc;
    bnp[128 + tid] = be - sc * mu;
  }
}

// ---------------------------------------------------------------------------
// Kernel 3: BN + sigmoid/elu + gated segment softmax pool (one wave per node)
//   self_fea_idx is sorted -> contiguous segments found by binary search;
//   online softmax handles any segment length in one pass.
// ---------------------------------------------------------------------------
__global__ void __launch_bounds__(256)
k_pool(const __bf16* __restrict__ zf, const __bf16* __restrict__ zc,
       const float* __restrict__ bnp, const float* __restrict__ Wg,
       const float* __restrict__ bgp, const float* __restrict__ aw,
       const int* __restrict__ selfi, const int* __restrict__ nbri,
       float* __restrict__ out) {
  const int lane = threadIdx.x & 31;
  const int node = blockIdx.x * WPB + (threadIdx.x >> 5);
  if (node >= NN) return;

  int lo = 0, hi = MM;
  while (lo < hi) { int mid = (lo + hi) >> 1; if (selfi[mid] < node) lo = mid + 1; else hi = mid; }
  const int start = lo;
  hi = MM;
  while (lo < hi) { int mid = (lo + hi) >> 1; if (selfi[mid] <= node) lo = mid + 1; else hi = mid; }
  const int end = lo;

  const float scf0 = bnp[lane],       shf0 = bnp[128 + lane];
  const float scf1 = bnp[lane + 32],  shf1 = bnp[160 + lane];
  const float scc0 = bnp[64 + lane],  shc0 = bnp[192 + lane];
  const float scc1 = bnp[96 + lane],  shc1 = bnp[224 + lane];
  const float wg0 = Wg[lane], wg1 = Wg[lane + 32];
  const float bg  = bgp[0];

  float mmax = -__builtin_inff(), denom = 0.f, acc0 = 0.f, acc1 = 0.f;
  for (int e = start; e < end; ++e) {
    const size_t base = (size_t)e * 64;
    float xf0 = scf0 * (float)zf[base + lane]      + shf0;
    float xf1 = scf1 * (float)zf[base + lane + 32] + shf1;
    float xc0 = scc0 * (float)zc[base + lane]      + shc0;
    float xc1 = scc1 * (float)zc[base + lane + 32] + shc1;
    float f0 = 1.0f / (1.0f + expf(-xf0));
    float f1 = 1.0f / (1.0f + expf(-xf1));
    float c0 = xc0 > 0.f ? xc0 : expm1f(xc0);
    float c1 = xc1 > 0.f ? xc1 : expm1f(xc1);
    float m0 = f0 * c0, m1 = f1 * c1;
    float p = m0 * wg0 + m1 * wg1;
#pragma unroll
    for (int off = 16; off > 0; off >>= 1) p += __shfl_xor(p, off, 32);
    float g = p + bg;
    float w = aw[nbri[e]];
    float nm   = fmaxf(mmax, g);
    float corr = expf(mmax - nm);      // 0 on first edge (mmax = -inf)
    float t    = w * expf(g - nm);
    denom = denom * corr + t;
    acc0  = acc0 * corr + t * m0;
    acc1  = acc1 * corr + t * m1;
    mmax  = nm;
  }
  float inv = 1.0f / (denom + 1e-13f);
  out[(size_t)node * 64 + lane]      = acc0 * inv;
  out[(size_t)node * 64 + lane + 32] = acc1 * inv;
}

// ---------------------------------------------------------------------------
extern "C" void kernel_launch(void* const* d_in, const int* in_sizes, int n_in,
                              void* d_out, int out_size, void* d_ws, size_t ws_size,
                              hipStream_t stream) {
  const float* aw    = (const float*)d_in[0];
  const float* atom  = (const float*)d_in[1];
  const float* bond  = (const float*)d_in[2];
  const float* Wf    = (const float*)d_in[3];
  /* d_in[4] = bf : cancels inside BatchNorm */
  const float* gf    = (const float*)d_in[5];
  const float* betaf = (const float*)d_in[6];
  const float* Wc    = (const float*)d_in[7];
  /* d_in[8] = bc : cancels inside BatchNorm */
  const float* gc    = (const float*)d_in[9];
  const float* betac = (const float*)d_in[10];
  const float* Wg    = (const float*)d_in[11];
  const float* bg    = (const float*)d_in[12];
  const int*   selfi = (const int*)d_in[13];
  const int*   nbri  = (const int*)d_in[14];
  float* out = (float*)d_out;

  char* ws = (char*)d_ws;
  size_t off = 0;
  __bf16* zf = (__bf16*)(ws + off);  off += (size_t)MM * 64 * 2;      // 160 MB
  __bf16* zc = (__bf16*)(ws + off);  off += (size_t)MM * 64 * 2;      // 160 MB
  __bf16* Wt = (__bf16*)(ws + off);  off += (size_t)128 * DINC * 2;   // 48 KB
  float* partial  = (float*)(ws + off);  off += (size_t)NB1 * 256 * 4; // ~10 MB
  float* partial2 = (float*)(ws + off);  off += (size_t)256 * 256 * 4;
  float* bnp      = (float*)(ws + off);

  k_prep<<<1, 256, 0, stream>>>(Wf, Wc, Wt);
  k_gemm<<<NB1, 256, 0, stream>>>(atom, bond, selfi, nbri, Wt, zf, zc, partial);
  k_red1<<<256, 256, 0, stream>>>(partial, partial2);
  k_red2<<<1, 256, 0, stream>>>(partial2, gf, betaf, gc, betac, bnp);
  k_pool<<<NN / WPB, 256, 0, stream>>>(zf, zc, bnp, Wg, bg, aw, selfi, nbri, out);
}
